// DenoisingTransformerLayer_19507741458644
// MI455X (gfx1250) — compile-verified, matrix operand order checked by reading, not run on hardware
//
#include <hip/hip_runtime.h>
#include <hip/hip_bf16.h>

#define N_NODES 50000
#define N_EDGES 800000
#define IN_DIM  128
#define HEADS   8
#define DHEAD   16
#define HD      128      // HEADS*DHEAD
#define CLAMPV  5.0f

typedef __attribute__((ext_vector_type(16))) __bf16   v16bf;
typedef __attribute__((ext_vector_type(8)))  float    v8f;
typedef __attribute__((ext_vector_type(8)))  unsigned v8u;

// ---------------- helpers ----------------

// pack two f32 into one dword holding two rne-rounded bf16 (lo -> [15:0])
__device__ __forceinline__ unsigned pk2bf(float lo, float hi) {
    unsigned ul = __float_as_uint(lo);
    ul = (ul + 0x7fffu + ((ul >> 16) & 1u)) >> 16;
    unsigned uh = __float_as_uint(hi);
    uh = (uh + 0x7fffu + ((uh >> 16) & 1u)) & 0xffff0000u;
    return uh | ul;
}

// monotone order-preserving float->uint key (for atomicMax-based segment max)
__device__ __forceinline__ unsigned fkey(float f) {
    unsigned u = __float_as_uint(f);
    return (u & 0x80000000u) ? ~u : (u | 0x80000000u);
}
__device__ __forceinline__ float keyf(unsigned k) {
    unsigned u = (k & 0x80000000u) ? (k & 0x7FFFFFFFu) : ~k;
    return __uint_as_float(u);
}

// load 16x32 bf16 A-fragment for one wave from a row-major f32 matrix.
// ISA layout (CDNA5 ISA §7.12.2): lane = {m = lane&15, khalf = lane>>4};
// elems 0..7 -> K = khalf*8 + i, elems 8..15 -> K = 16 + khalf*8 + (i-8).
__device__ __forceinline__ v16bf load_a_frag(const float* __restrict__ rowp, int kc, int khalf) {
    int k0 = kc * 32 + khalf * 8;
    float4 f0 = *(const float4*)(rowp + k0);
    float4 f1 = *(const float4*)(rowp + k0 + 4);
    float4 f2 = *(const float4*)(rowp + k0 + 16);
    float4 f3 = *(const float4*)(rowp + k0 + 20);
    v8u t;
    t[0] = pk2bf(f0.x, f0.y);  t[1] = pk2bf(f0.z, f0.w);
    t[2] = pk2bf(f1.x, f1.y);  t[3] = pk2bf(f1.z, f1.w);
    t[4] = pk2bf(f2.x, f2.y);  t[5] = pk2bf(f2.z, f2.w);
    t[6] = pk2bf(f3.x, f3.y);  t[7] = pk2bf(f3.z, f3.w);
    return __builtin_bit_cast(v16bf, t);
}

__device__ __forceinline__ v8f wmma_bf16(v16bf a, v16bf b, v8f c) {
    return __builtin_amdgcn_wmma_f32_16x16x32_bf16(false, a, false, b, (short)0, c, false, false);
}

// ---------------- kernel 0: pack W{Q,K,V,E1} (f32 row-major 128x128) into bf16
// per-lane B-fragment layout, one 32-byte v16bf blob per (mat,kc,nt,lane) ------
__global__ void pack_weights_kernel(const float* __restrict__ WQ, const float* __restrict__ WK,
                                    const float* __restrict__ WV, const float* __restrict__ WE1,
                                    unsigned* __restrict__ pW) {  // dword-typed: 8 dwords/frag
    int idx = blockIdx.x * blockDim.x + threadIdx.x;     // 4*4*8*32 = 4096 fragments
    if (idx >= 4 * 4 * 8 * 32) return;
    int lane = idx & 31;
    int nt   = (idx >> 5) & 7;
    int kc   = (idx >> 8) & 3;
    int mat  = idx >> 10;
    const float* W = (mat == 0) ? WQ : (mat == 1) ? WK : (mat == 2) ? WV : WE1;
    int n = nt * 16 + (lane & 15);
    int khalf = lane >> 4;
    unsigned* dst = pW + (size_t)idx * 8;
    #pragma unroll
    for (int j = 0; j < 8; ++j) {
        int i0 = 2 * j;                                  // fragment elements 2j, 2j+1
        int K0 = kc * 32 + ((i0 < 8) ? (khalf * 8 + i0) : (16 + khalf * 8 + (i0 - 8)));
        int i1 = 2 * j + 1;
        int K1 = kc * 32 + ((i1 < 8) ? (khalf * 8 + i1) : (16 + khalf * 8 + (i1 - 8)));
        dst[j] = pk2bf(W[K0 * HD + n], W[K1 * HD + n]);
    }
}

// ---------------- kernel 1: zero wV and softmax stats ----------------
__global__ void init_kernel(float* __restrict__ wV, float* __restrict__ denom,
                            unsigned* __restrict__ segmax) {
    int i = blockIdx.x * blockDim.x + threadIdx.x;
    if (i < N_NODES * HD) wV[i] = 0.0f;
    if (i < N_NODES * HEADS) { denom[i] = 0.0f; segmax[i] = 0u; }  // key 0 ~= -inf
}

// ---------------- kernel 2: node projections Q,K,V via bf16 WMMA ----------------
// one wave per 16-row tile; one matrix pass at a time (single accumulator ->
// no spills); A fragments stay resident across the 3 passes so x is read once.
__global__ void node_qkv_kernel(const float* __restrict__ x, const __bf16* __restrict__ pW,
                                const float* __restrict__ bQ,
                                float* __restrict__ Qb, float* __restrict__ Kb,
                                float* __restrict__ Vb) {
    int wave = threadIdx.x >> 5, lane = threadIdx.x & 31;
    int tile = blockIdx.x * (blockDim.x >> 5) + wave;
    if (tile >= N_NODES / 16) return;          // wave-uniform exit (EXEC all-1 at WMMA)
    int rowBase = tile * 16;
    int m = lane & 15, khalf = lane >> 4;
    const float* xr = x + (size_t)(rowBase + m) * IN_DIM;

    v16bf a[4];
    #pragma unroll
    for (int kc = 0; kc < 4; ++kc) a[kc] = load_a_frag(xr, kc, khalf);

    #pragma unroll
    for (int mat = 0; mat < 3; ++mat) {
        const __bf16* pWm = pW + (size_t)mat * 1024 * 16;
        float* outp = (mat == 0) ? Qb : (mat == 1) ? Kb : Vb;
        #pragma unroll
        for (int nt = 0; nt < 8; ++nt) {
            v8f acc = {};
            #pragma unroll
            for (int kc = 0; kc < 4; ++kc) {
                v16bf b = *(const v16bf*)(pWm + (size_t)((kc * 8 + nt) * 32 + lane) * 16);
                acc = wmma_bf16(a[kc], b, acc);
            }
            int col = nt * 16 + m;
            float bias = (mat == 0) ? bQ[col] : 0.0f;
            #pragma unroll
            for (int r = 0; r < 8; ++r) {
                int orow = rowBase + khalf * 8 + r;   // C-layout: VGPR r -> M=r (+8 hi half)
                outp[(size_t)orow * HD + col] = acc[r] + bias;
            }
        }
    }
}

// ---------------- kernel 3: edge projection E + score_e + wE + clamped per-head
// score + segment-max.  One wave per 16-edge tile.  N-tile nt == head index. ----
__global__ void edge_score_kernel(const float* __restrict__ edge_attr,
                                  const __bf16* __restrict__ pW,  // mat 3 at 3*1024*16
                                  const float* __restrict__ bE1,
                                  const int* __restrict__ eidx,
                                  const float* __restrict__ Qb, const float* __restrict__ Kb,
                                  float* __restrict__ wE, float* __restrict__ sc,
                                  unsigned* __restrict__ segmax) {
    int wave = threadIdx.x >> 5, lane = threadIdx.x & 31;
    int tile = blockIdx.x * (blockDim.x >> 5) + wave;
    if (tile >= N_EDGES / 16) return;
    int eBase = tile * 16;
    int m = lane & 15, khalf = lane >> 4;
    const float* er = edge_attr + (size_t)(eBase + m) * IN_DIM;

    // prefetch next tile's rows of edge_attr (gfx1250 global_prefetch_b8 path)
    {
        int pf = eBase + m + 64; if (pf >= N_EDGES) pf = N_EDGES - 1;
        __builtin_prefetch(edge_attr + (size_t)pf * IN_DIM, 0, 1);
    }

    v16bf a[4];
    #pragma unroll
    for (int kc = 0; kc < 4; ++kc) a[kc] = load_a_frag(er, kc, khalf);

    // the 8 edges this lane's C-fragment rows correspond to
    int srcr[8], dstr[8];
    #pragma unroll
    for (int r = 0; r < 8; ++r) {
        int e = eBase + khalf * 8 + r;
        srcr[r] = eidx[e];
        dstr[r] = eidx[N_EDGES + e];
    }

    const __bf16* pWE = pW + (size_t)3 * 1024 * 16;
    #pragma unroll
    for (int nt = 0; nt < 8; ++nt) {          // nt is also the head index (16 cols/head)
        v8f acc = {};
        #pragma unroll
        for (int kc = 0; kc < 4; ++kc) {
            v16bf b = *(const v16bf*)(pWE + (size_t)((kc * 8 + nt) * 32 + lane) * 16);
            acc = wmma_bf16(a[kc], b, acc);
        }
        int col = nt * 16 + m;
        float bias = bE1[col];
        float s[8];
        #pragma unroll
        for (int r = 0; r < 8; ++r) {
            float Ev = acc[r] + bias;
            float kv = Kb[(size_t)srcr[r] * HD + col];
            float qv = Qb[(size_t)dstr[r] * HD + col];
            float sv = kv * qv * Ev;                      // score_e
            wE[(size_t)(eBase + khalf * 8 + r) * HD + col] = sv;
            s[r] = sv;
        }
        // per-head (sum over D=16 columns) reduction within each 16-lane half
        #pragma unroll
        for (int r = 0; r < 8; ++r) {
            float red = s[r];
            red += __shfl_xor(red, 1, 16);
            red += __shfl_xor(red, 2, 16);
            red += __shfl_xor(red, 4, 16);
            red += __shfl_xor(red, 8, 16);
            if (m == r) {
                int e = eBase + khalf * 8 + r;
                float sh = red * 0.25f;                   // / sqrt(D), D=16
                sh = fminf(CLAMPV, fmaxf(-CLAMPV, sh));
                sc[(size_t)e * HEADS + nt] = sh;
                atomicMax(&segmax[(size_t)dstr[r] * HEADS + nt], fkey(sh));
            }
        }
    }
}

// ---------------- kernel 4: ex = exp(score - segmax[dst]); denom += ex ----------
__global__ void softmax_denom_kernel(const int* __restrict__ eidx, float* __restrict__ sc,
                                     const unsigned* __restrict__ segmax,
                                     float* __restrict__ denom) {
    int i = blockIdx.x * blockDim.x + threadIdx.x;     // N_EDGES*HEADS
    if (i >= N_EDGES * HEADS) return;
    int e = i >> 3, h = i & 7;
    int dst = eidx[N_EDGES + e];
    float mx = keyf(segmax[dst * HEADS + h]);
    float ex = __expf(sc[i] - mx);
    sc[i] = ex;                                        // reuse buffer: score -> ex
    atomicAdd(&denom[dst * HEADS + h], ex);
}

// ---------------- kernel 5: wV[dst] += (V[src] + score_e) * alpha ----------------
__global__ void aggregate_kernel(const int* __restrict__ eidx, const float* __restrict__ sc,
                                 const float* __restrict__ denom, const float* __restrict__ Vb,
                                 const float* __restrict__ wE, float* __restrict__ wV) {
    int i = blockIdx.x * blockDim.x + threadIdx.x;     // N_EDGES*HD = 102.4M (< 2^31)
    if (i >= N_EDGES * HD) return;
    int e = i >> 7, col = i & 127, h = col >> 4;
    int src = eidx[e], dst = eidx[N_EDGES + e];
    float alpha = sc[e * HEADS + h] / (denom[dst * HEADS + h] + 1e-16f);
    float val = (Vb[(size_t)src * HD + col] + wE[i]) * alpha;
    atomicAdd(&wV[(size_t)dst * HD + col], val);
}

// ---------------- launch ----------------
extern "C" void kernel_launch(void* const* d_in, const int* in_sizes, int n_in,
                              void* d_out, int out_size, void* d_ws, size_t ws_size,
                              hipStream_t stream) {
    const float* x         = (const float*)d_in[0];
    const float* edge_attr = (const float*)d_in[1];
    const int*   eidx      = (const int*)d_in[2];
    const float* WQ        = (const float*)d_in[3];
    const float* bQ        = (const float*)d_in[4];
    const float* WK        = (const float*)d_in[5];
    const float* WV        = (const float*)d_in[6];
    const float* WE1       = (const float*)d_in[7];
    const float* bE1       = (const float*)d_in[8];

    float* wV = (float*)d_out;                         // [N_NODES*128]
    float* wE = wV + (size_t)N_NODES * HD;             // [N_EDGES*128]

    // workspace layout (~106 MB)
    char* ws = (char*)d_ws;
    __bf16*   pW     = (__bf16*)ws;                                   // 4*128*128 bf16 = 128 KB
    float*    Qb     = (float*)(ws + 131072);                         // 25.6 MB
    float*    Kb     = Qb + (size_t)N_NODES * HD;
    float*    Vb     = Kb + (size_t)N_NODES * HD;
    float*    sc     = Vb + (size_t)N_NODES * HD;                     // N_EDGES*8 f32
    float*    denom  = sc + (size_t)N_EDGES * HEADS;                  // N_NODES*8
    unsigned* segmax = (unsigned*)(denom + (size_t)N_NODES * HEADS);  // N_NODES*8

    pack_weights_kernel<<<(4096 + 255) / 256, 256, 0, stream>>>(WQ, WK, WV, WE1, (unsigned*)pW);

    int initN = N_NODES * HD;
    init_kernel<<<(initN + 255) / 256, 256, 0, stream>>>(wV, denom, segmax);

    int nodeTiles = N_NODES / 16;                      // 3125
    node_qkv_kernel<<<(nodeTiles + 3) / 4, 128, 0, stream>>>(x, pW, bQ, Qb, Kb, Vb);

    int edgeTiles = N_EDGES / 16;                      // 50000
    edge_score_kernel<<<(edgeTiles + 3) / 4, 128, 0, stream>>>(edge_attr, pW, bE1, eidx,
                                                               Qb, Kb, wE, sc, segmax);

    int ehN = N_EDGES * HEADS;
    softmax_denom_kernel<<<(ehN + 255) / 256, 256, 0, stream>>>(eidx, sc, segmax, denom);

    int aggN = N_EDGES * HD;
    aggregate_kernel<<<(aggN + 255) / 256, 256, 0, stream>>>(eidx, sc, denom, Vb, wE, wV);
}
